// Qwen3OmniVisionEncoder_27711128994365
// MI455X (gfx1250) — matrix operand order repaired and checked by reading, not compiled, not run on hardware
//
#include <hip/hip_runtime.h>

// ---------------------------------------------------------------------------
// Qwen3-Omni vision encoder block, MI455X (gfx1250, wave32, WMMA bf16).
// Compute-bound (~204 GFLOP, weights fit in 192MB L2) -> everything matmul-
// shaped runs on v_wmma_f32_16x16x32_bf16 with f32 accumulation.
// GEMM A-tile staging uses gfx1250 async global->LDS copies (ASYNCcnt).
// ---------------------------------------------------------------------------

#define SEQ 2048
#define DIM 1280
#define NH  16
#define HDM 80          // head dim
#define FF  5120
#define NLAYER 2

typedef __attribute__((ext_vector_type(16))) __bf16 v16bf;
typedef __attribute__((ext_vector_type(8)))  float  v8f;

union Frag { v16bf v; unsigned int u[8]; };

__device__ inline unsigned short f2bf(float f) {
    unsigned int u = __float_as_uint(f);
    unsigned int r = u + 0x7FFFu + ((u >> 16) & 1u);   // round-to-nearest-even
    return (unsigned short)(r >> 16);
}

__device__ inline v8f zero8() {
    v8f v;
#pragma unroll
    for (int i = 0; i < 8; ++i) v[i] = 0.0f;
    return v;
}

__device__ inline float redmax16(float v) {
#pragma unroll
    for (int m = 1; m < 16; m <<= 1) v = fmaxf(v, __shfl_xor(v, m, 32));
    return v;
}
__device__ inline float redsum16(float v) {
#pragma unroll
    for (int m = 1; m < 16; m <<= 1) v += __shfl_xor(v, m, 32);
    return v;
}

// ---------------------------------------------------------------------------
// fp32 -> bf16 conversion (weights, per-use)
// ---------------------------------------------------------------------------
__global__ __launch_bounds__(256) void cvt_bf16_kernel(const float* __restrict__ in,
                                                       unsigned short* __restrict__ out,
                                                       int n) {
    int i = blockIdx.x * blockDim.x + threadIdx.x;
    int stride = gridDim.x * blockDim.x;
    for (; i < n; i += stride) out[i] = f2bf(in[i]);
}

// ---------------------------------------------------------------------------
// LayerNorm: fp32 in -> bf16 normalized out (one row per block)
// ---------------------------------------------------------------------------
__global__ __launch_bounds__(256) void layernorm_kernel(const float* __restrict__ x,
                                                        const float* __restrict__ g,
                                                        const float* __restrict__ b,
                                                        unsigned short* __restrict__ out) {
    __shared__ float s1[256];
    __shared__ float s2[256];
    const int s = blockIdx.x, t = threadIdx.x;
    const float* row = x + (size_t)s * DIM;
    float a = 0.f, a2 = 0.f;
    for (int i = t; i < DIM; i += 256) { float v = row[i]; a += v; a2 += v * v; }
    s1[t] = a; s2[t] = a2;
    __syncthreads();
    for (int o = 128; o > 0; o >>= 1) {
        if (t < o) { s1[t] += s1[t + o]; s2[t] += s2[t + o]; }
        __syncthreads();
    }
    float mean = s1[0] * (1.0f / DIM);
    float var  = s2[0] * (1.0f / DIM) - mean * mean;
    float inv  = rsqrtf(var + 1e-6f);
    for (int i = t; i < DIM; i += 256) {
        float v = (row[i] - mean) * inv * g[i] + b[i];
        out[(size_t)s * DIM + i] = f2bf(v);
    }
}

// ---------------------------------------------------------------------------
// GEMM: C[M,N] = A[M,K](bf16) * B[K,N](bf16) + bias (+resid) (+gelu)
// WG = 256 threads = 8 waves; tile 128x128; wave tile 64x32 (4x2 WMMA tiles).
// A tile staged with global_load_async_to_lds_b128 (bypasses VGPRs, ASYNCcnt);
// B tile staged manually because it is transposed to [n][k] so B-fragment
// bf16 pairs are contiguous 32-bit LDS reads matching the WMMA VGPR layouts.
// ---------------------------------------------------------------------------
__global__ __launch_bounds__(256) void gemm_bf16_kernel(
    const unsigned short* __restrict__ A, const unsigned short* __restrict__ Bw,
    const float* __restrict__ bias, const float* __restrict__ resid,
    float* __restrict__ outF, unsigned short* __restrict__ outB,
    int M, int K, int N, int act) {
    __shared__ __align__(16) unsigned short lA[128][32];
    __shared__ __align__(16) unsigned short lB[128][32];

    const int tid  = threadIdx.x;
    const int lane = tid & 31;
    const int wave = tid >> 5;
    const int wm = wave & 1;        // 2 waves along M
    const int wn = wave >> 1;       // 4 waves along N
    const int bm = blockIdx.x * 128;
    const int bn = blockIdx.y * 128;
    const int hi = (lane < 16) ? 0 : 1;
    const int lr = lane & 15;

    v8f acc[4][2];
#pragma unroll
    for (int i = 0; i < 4; ++i)
#pragma unroll
        for (int j = 0; j < 2; ++j) acc[i][j] = zero8();

    const int arow = tid >> 1;           // 0..127
    const int akt  = (tid & 1) * 16;     // 0 / 16
    const int bk   = tid >> 3;           // 0..31
    const int bn0  = (tid & 7) * 16;     // 0..112

    // LDS byte offset of this thread's A-tile slice (generic LDS pointer:
    // low 32 bits are the LDS offset)
    const unsigned int ldsA = (unsigned int)(unsigned long long)(&lA[arow][akt]);

    for (int k0 = 0; k0 < K; k0 += 32) {
        // stage A tile: 32B per thread via two async global->LDS b128 copies.
        // offset:16 applies to both the global and LDS side (ISA: dsaddr and
        // memaddr both add INST_OFFSET).
        {
            const unsigned long long ga =
                (unsigned long long)(A + (size_t)(bm + arow) * K + k0 + akt);
            asm volatile(
                "global_load_async_to_lds_b128 %0, %1, off\n\t"
                "global_load_async_to_lds_b128 %0, %1, off offset:16"
                :: "v"(ldsA), "v"(ga) : "memory");
        }
        // stage B tile transposed: lB[n][k] = B[k0+k][bn+n]
        {
            const unsigned short* src = Bw + (size_t)(k0 + bk) * N + bn + bn0;
#pragma unroll
            for (int i = 0; i < 16; ++i) lB[bn0 + i][bk] = src[i];
        }
        if (k0 + 32 < K) {   // gfx1250 global_prefetch_b8 of next K tile
            __builtin_prefetch(A + (size_t)(bm + arow) * K + k0 + 32 + akt, 0, 1);
            __builtin_prefetch(Bw + (size_t)(k0 + 32 + bk) * N + bn + bn0, 0, 1);
        }
        // own async copies complete, then cross-wave barrier
        asm volatile("s_wait_asynccnt 0x0" ::: "memory");
        __syncthreads();

        Frag fa[4];
#pragma unroll
        for (int mt = 0; mt < 4; ++mt) {
            const int m = wm * 64 + mt * 16 + lr;
#pragma unroll
            for (int j = 0; j < 8; ++j) {
                const int k = 2 * (j & 3) + ((j >= 4) ? 16 : 0) + (hi ? 8 : 0);
                fa[mt].u[j] = *(const unsigned int*)&lA[m][k];
            }
        }
#pragma unroll
        for (int nt = 0; nt < 2; ++nt) {
            Frag fb;
            const int n = wn * 32 + nt * 16 + lr;
#pragma unroll
            for (int j = 0; j < 8; ++j) {
                const int k = 2 * (j & 3) + ((j >= 4) ? 16 : 0) + (hi ? 8 : 0);
                fb.u[j] = *(const unsigned int*)&lB[n][k];
            }
#pragma unroll
            for (int mt = 0; mt < 4; ++mt) {
                acc[mt][nt] = __builtin_amdgcn_wmma_f32_16x16x32_bf16(
                    false, fa[mt].v, false, fb.v, (short)0, acc[mt][nt], false, false);
            }
        }
        __syncthreads();
    }

    // epilogue: bias / residual / exact gelu; write f32 and/or bf16
#pragma unroll
    for (int mt = 0; mt < 4; ++mt)
#pragma unroll
        for (int nt = 0; nt < 2; ++nt)
#pragma unroll
            for (int j = 0; j < 8; ++j) {
                const int row = bm + wm * 64 + mt * 16 + j + (hi ? 8 : 0);
                const int col = bn + wn * 32 + nt * 16 + lr;
                float v = acc[mt][nt][j] + bias[col];
                if (resid) v += resid[(size_t)row * N + col];
                if (act == 1) v = 0.5f * v * (1.0f + erff(v * 0.70710678118654752f));
                if (outF) outF[(size_t)row * N + col] = v;
                if (outB) outB[(size_t)row * N + col] = f2bf(v);
            }
}

// ---------------------------------------------------------------------------
// RoPE + repack:  qkv[S,3*D] f32 -> qh[H][S][80] (scaled by 80^-0.5, bf16),
// kh[H][S][80] bf16, vt[H][80][S] bf16 (transposed so PV B-fragments load
// contiguous key pairs).
// ---------------------------------------------------------------------------
__global__ __launch_bounds__(256) void rope_pack_kernel(
    const float* __restrict__ qkv, const float* __restrict__ cs, const float* __restrict__ sn,
    unsigned short* __restrict__ qh, unsigned short* __restrict__ kh,
    unsigned short* __restrict__ vt) {
    const int s = blockIdx.x, t = threadIdx.x;
    const float* base = qkv + (size_t)s * (3 * DIM);
    for (int idx = t; idx < DIM; idx += 256) {
        const int h = idx / HDM, d = idx % HDM;
        const float c = cs[s * HDM + d], si = sn[s * HDM + d];
        const int pidx = (d < 40) ? idx + 40 : idx - 40;
        const float sgn = (d < 40) ? -1.f : 1.f;
        const float q = base[idx], k = base[DIM + idx], v = base[2 * DIM + idx];
        const float qr = sgn * base[pidx], kr = sgn * base[DIM + pidx];
        const float qo = q * c + qr * si;
        const float ko = k * c + kr * si;
        qh[((size_t)h * SEQ + s) * HDM + d] = f2bf(qo * 0.11180339887498949f); // * 80^-0.5
        kh[((size_t)h * SEQ + s) * HDM + d] = f2bf(ko);
        vt[((size_t)h * HDM + d) * SEQ + s] = f2bf(v);
    }
}

// ---------------------------------------------------------------------------
// Flash attention: one wave (32 threads) per (head, 16-query tile).
// QK^T: 3 WMMAs (hd=80 zero-padded to 96); online softmax with 16-lane shfl
// row reductions (row stats stay in registers: C-layout row m = j + 8*hi);
// P redistributed C->A layout via 1KB LDS; PV: 5 WMMAs per 32-key chunk.
// ---------------------------------------------------------------------------
__global__ __launch_bounds__(32) void attn_kernel(
    const unsigned short* __restrict__ qh, const unsigned short* __restrict__ kh,
    const unsigned short* __restrict__ vt, unsigned short* __restrict__ outB) {
    __shared__ __align__(16) unsigned short lP[16][32];
    const int lane = threadIdx.x;
    const int h  = blockIdx.y;
    const int q0 = blockIdx.x * 16;
    const int hi = (lane < 16) ? 0 : 1;
    const int lr = lane & 15;

    // Q as A-fragments (3 chunks of K=32, last zero-padded past d=80)
    Frag aQ[3];
#pragma unroll
    for (int c = 0; c < 3; ++c)
#pragma unroll
        for (int j = 0; j < 8; ++j) {
            const int k = 2 * (j & 3) + ((j >= 4) ? 16 : 0) + (hi ? 8 : 0);
            const int d = c * 32 + k;
            unsigned int u = 0;
            if (d < HDM)
                u = *(const unsigned int*)(qh + ((size_t)h * SEQ + (q0 + lr)) * HDM + d);
            aQ[c].u[j] = u;
        }

    v8f accO[5];
#pragma unroll
    for (int c = 0; c < 5; ++c) accO[c] = zero8();
    float mI[8], lI[8];
#pragma unroll
    for (int j = 0; j < 8; ++j) { mI[j] = -1e30f; lI[j] = 0.f; }

    for (int kb = 0; kb < SEQ / 32; ++kb) {
        const int key0 = kb * 32;
        v8f sc0 = zero8(), sc1 = zero8();
#pragma unroll
        for (int t = 0; t < 2; ++t) {
            v8f sc = zero8();
#pragma unroll
            for (int c = 0; c < 3; ++c) {
                Frag bK;
#pragma unroll
                for (int j = 0; j < 8; ++j) {
                    const int k = 2 * (j & 3) + ((j >= 4) ? 16 : 0) + (hi ? 8 : 0);
                    const int d = c * 32 + k;
                    unsigned int u = 0;
                    if (d < HDM) {
                        const int key = key0 + t * 16 + lr;
                        u = *(const unsigned int*)(kh + ((size_t)h * SEQ + key) * HDM + d);
                    }
                    bK.u[j] = u;
                }
                sc = __builtin_amdgcn_wmma_f32_16x16x32_bf16(
                    false, aQ[c].v, false, bK.v, (short)0, sc, false, false);
            }
            if (t == 0) sc0 = sc; else sc1 = sc;
        }

        // online softmax (row m = j + 8*hi; 16 lanes of a half-wave hold the row)
        float fac[8], p0[8], p1[8];
#pragma unroll
        for (int j = 0; j < 8; ++j) {
            float vmax = redmax16(fmaxf(sc0[j], sc1[j]));
            const float mnew = fmaxf(mI[j], vmax);
            const float f = __expf(mI[j] - mnew);
            const float e0 = __expf(sc0[j] - mnew);
            const float e1 = __expf(sc1[j] - mnew);
            lI[j] = lI[j] * f + redsum16(e0 + e1);
            mI[j] = mnew;
            fac[j] = f; p0[j] = e0; p1[j] = e1;
        }
#pragma unroll
        for (int c = 0; c < 5; ++c)
#pragma unroll
            for (int j = 0; j < 8; ++j) accO[c][j] *= fac[j];

        // redistribute P: C-layout -> A-layout via LDS
#pragma unroll
        for (int j = 0; j < 8; ++j) {
            const int r = j + (hi ? 8 : 0);
            lP[r][lr]      = f2bf(p0[j]);
            lP[r][lr + 16] = f2bf(p1[j]);
        }
        __syncthreads();
        Frag aP;
#pragma unroll
        for (int j = 0; j < 8; ++j) {
            const int k = 2 * (j & 3) + ((j >= 4) ? 16 : 0) + (hi ? 8 : 0);
            aP.u[j] = *(const unsigned int*)&lP[lr][k];
        }
        // P @ V : 5 output-dim chunks of 16
#pragma unroll
        for (int c = 0; c < 5; ++c) {
            Frag bV;
#pragma unroll
            for (int j = 0; j < 8; ++j) {
                const int k = 2 * (j & 3) + ((j >= 4) ? 16 : 0) + (hi ? 8 : 0);
                bV.u[j] = *(const unsigned int*)(vt + ((size_t)h * HDM + (c * 16 + lr)) * SEQ + key0 + k);
            }
            accO[c] = __builtin_amdgcn_wmma_f32_16x16x32_bf16(
                false, aP.v, false, bV.v, (short)0, accO[c], false, false);
        }
        __syncthreads();
    }

#pragma unroll
    for (int c = 0; c < 5; ++c)
#pragma unroll
        for (int j = 0; j < 8; ++j) {
            const int m = j + (hi ? 8 : 0);
            const float v = accO[c][j] / lI[j];
            outB[(size_t)(q0 + m) * DIM + h * HDM + c * 16 + lr] = f2bf(v);
        }
}

// ---------------------------------------------------------------------------
// Host orchestration
// ---------------------------------------------------------------------------
extern "C" void kernel_launch(void* const* d_in, const int* in_sizes, int n_in,
                              void* d_out, int out_size, void* d_ws, size_t ws_size,
                              hipStream_t stream) {
    const float* hs    = (const float*)d_in[0];
    const float* cosT  = (const float*)d_in[1];
    const float* sinT  = (const float*)d_in[2];
    const float* qkvW  = (const float*)d_in[3];
    const float* qkvB  = (const float*)d_in[4];
    const float* projW = (const float*)d_in[5];
    const float* projB = (const float*)d_in[6];
    const float* ln1g  = (const float*)d_in[7];
    const float* ln1b  = (const float*)d_in[8];
    const float* ln2g  = (const float*)d_in[9];
    const float* ln2b  = (const float*)d_in[10];
    const float* fc1W  = (const float*)d_in[11];
    const float* fc1B  = (const float*)d_in[12];
    const float* fc2W  = (const float*)d_in[13];
    const float* fc2B  = (const float*)d_in[14];

    float* x = (float*)d_out;   // residual stream lives in d_out

    char* base = (char*)d_ws;
    size_t off = 0;
    auto take = [&](size_t bytes) -> void* {
        void* p = base + off;
        off += (bytes + 255) & ~(size_t)255;
        return p;
    };
    unsigned short* hnorm = (unsigned short*)take((size_t)SEQ * DIM * 2);       // ln output bf16
    unsigned short* wB    = (unsigned short*)take((size_t)FF * DIM * 2);        // bf16 weight staging
    float*          qkvF  = (float*)take((size_t)SEQ * 3 * DIM * 4);            // qkv f32
    unsigned short* qhB   = (unsigned short*)take((size_t)NH * SEQ * HDM * 2);  // roped q (scaled)
    unsigned short* khB   = (unsigned short*)take((size_t)NH * SEQ * HDM * 2);  // roped k
    unsigned short* vtB   = (unsigned short*)take((size_t)NH * HDM * SEQ * 2);  // v transposed
    unsigned short* attnB = (unsigned short*)take((size_t)SEQ * DIM * 2);       // attention out bf16
    unsigned short* mlpB  = (unsigned short*)take((size_t)SEQ * FF * 2);        // gelu(fc1) bf16
    (void)ws_size; (void)in_sizes; (void)n_in; (void)out_size;

    hipMemcpyAsync(x, hs, (size_t)SEQ * DIM * sizeof(float),
                   hipMemcpyDeviceToDevice, stream);

    for (int l = 0; l < NLAYER; ++l) {
        // ---- attention ----
        layernorm_kernel<<<SEQ, 256, 0, stream>>>(x, ln1g + l * DIM, ln1b + l * DIM, hnorm);
        cvt_bf16_kernel<<<2048, 256, 0, stream>>>(qkvW + (size_t)l * DIM * 3 * DIM, wB, DIM * 3 * DIM);
        gemm_bf16_kernel<<<dim3(SEQ / 128, (3 * DIM) / 128), 256, 0, stream>>>(
            hnorm, wB, qkvB + (size_t)l * 3 * DIM, nullptr, qkvF, nullptr, SEQ, DIM, 3 * DIM, 0);
        rope_pack_kernel<<<SEQ, 256, 0, stream>>>(qkvF, cosT, sinT, qhB, khB, vtB);
        attn_kernel<<<dim3(SEQ / 16, NH), 32, 0, stream>>>(qhB, khB, vtB, attnB);
        cvt_bf16_kernel<<<2048, 256, 0, stream>>>(projW + (size_t)l * DIM * DIM, wB, DIM * DIM);
        gemm_bf16_kernel<<<dim3(SEQ / 128, DIM / 128), 256, 0, stream>>>(
            attnB, wB, projB + (size_t)l * DIM, x, x, nullptr, SEQ, DIM, DIM, 0);
        // ---- MLP ----
        layernorm_kernel<<<SEQ, 256, 0, stream>>>(x, ln2g + l * DIM, ln2b + l * DIM, hnorm);
        cvt_bf16_kernel<<<2048, 256, 0, stream>>>(fc1W + (size_t)l * DIM * FF, wB, DIM * FF);
        gemm_bf16_kernel<<<dim3(SEQ / 128, FF / 128), 256, 0, stream>>>(
            hnorm, wB, fc1B + (size_t)l * FF, nullptr, nullptr, mlpB, SEQ, DIM, FF, 1);
        cvt_bf16_kernel<<<2048, 256, 0, stream>>>(fc2W + (size_t)l * FF * DIM, wB, FF * DIM);
        gemm_bf16_kernel<<<dim3(SEQ / 128, DIM / 128), 256, 0, stream>>>(
            mlpB, wB, fc2B + (size_t)l * DIM, x, x, nullptr, SEQ, FF, DIM, 0);
    }
}
